// EncoderMach5_60979945669357
// MI455X (gfx1250) — matrix operand order repaired and checked by reading, not compiled, not run on hardware
//
#include <hip/hip_runtime.h>
#include <hip/hip_fp16.h>

// CapsNet encoder on MI455X (gfx1250, wave32).
// GEMM-shaped work (conv1 / primarycaps im2col GEMMs, decoder MLP) runs on
// v_wmma_f32_16x16x32_f16 with double-buffered async global->LDS staging
// (GLOBAL_LOAD_ASYNC_TO_LDS_B128 + s_wait_asynccnt). EM routing is VALU with
// LDS-resident R.

typedef __attribute__((ext_vector_type(16))) _Float16 v16h;
typedef __attribute__((ext_vector_type(8)))  _Float16 v8h;
typedef __attribute__((ext_vector_type(8)))  float    v8f;
typedef __attribute__((ext_vector_type(4)))  int      v4i;

#define TPB 256
#define CAPS_EPS 1e-8f
#define BNEPS 1e-3f
#define KC 128                 // K-chunk staged per LDS buffer (halves)

#if defined(__gfx1250__) && __has_builtin(__builtin_amdgcn_global_load_async_to_lds_b128)
#define ASYNC_OK 1
#else
#define ASYNC_OK 0
#endif

__device__ __forceinline__ void async_cp16(const __half* g, __half* l) {
#if ASYNC_OK
  __builtin_amdgcn_global_load_async_to_lds_b128(
      (__attribute__((address_space(1))) v4i*)g,
      (__attribute__((address_space(3))) v4i*)l, 0, 0);
#else
  *(v8h*)l = *(const v8h*)g;   // fallback: vmem load + ds_store
#endif
}

__device__ __forceinline__ void async_wait_keep6() {
#if ASYNC_OK
  asm volatile("s_wait_asynccnt 0x6" ::: "memory");
#endif
}
__device__ __forceinline__ void async_wait_all() {
#if ASYNC_OK
  asm volatile("s_wait_asynccnt 0x0" ::: "memory");
#endif
}

// ---------------------------------------------------------------- WMMA GEMM
// A: [M,K] row-major f16 (M%64==0, K%128==0). Bt: [N,K] f16 (B transposed,
// so each lane's B fragment is one contiguous 32B run). C: [M,N] f32, N%32==0.
// Block: 8 waves -> 64x32 macro-tile; K double-buffered through LDS.

__device__ __forceinline__ v16h load_a_frag(const __half* row, int hi) {
  // A 16x32 f16 layout (matches B): lanes 0-15 halves0..15=K0..15;
  // lanes 16-31 halves0..15=K16..31.
  return *(const v16h*)(row + hi * 16);
}

__global__ __launch_bounds__(TPB) void k_gemm_wmma(const __half* __restrict__ A,
    const __half* __restrict__ Bt, float* __restrict__ C, int M, int N, int K) {
  __shared__ __half smA[2][64 * KC];   // 32 KB
  __shared__ __half smB[2][32 * KC];   // 16 KB
  const int t = threadIdx.x;
  const int lane = t & 31;
  const int wave = t >> 5;
  const int nBlk = N >> 5;
  const int bm = (int)blockIdx.x / nBlk;
  const int bn = (int)blockIdx.x - bm * nBlk;
  const int rowBase = bm * 64;
  const int colBase = bn * 32;
  const int tmL = wave & 3, tnL = wave >> 2;
  const int r = lane & 15, hi = lane >> 4;

  // stage one KC-chunk of the A (64 rows) and B (32 rows) slabs:
  // 6 async b128 per thread -> 6 per-wave ASYNCcnt increments per buffer.
  auto stage = [&](int buf, int k0) {
    #pragma unroll
    for (int i = 0; i < 4; ++i) {
      int idx = t + i * 256;                 // 1024 chunks of 8 halves
      int row = idx >> 4, seg = idx & 15;
      async_cp16(A + (size_t)(rowBase + row) * K + k0 + seg * 8,
                 &smA[buf][row * KC + seg * 8]);
    }
    #pragma unroll
    for (int i = 0; i < 2; ++i) {
      int idx = t + i * 256;                 // 512 chunks of 8 halves
      int row = idx >> 4, seg = idx & 15;
      async_cp16(Bt + (size_t)(colBase + row) * K + k0 + seg * 8,
                 &smB[buf][row * KC + seg * 8]);
    }
  };

  v8f acc = {};
  const int nch = K / KC;
  stage(0, 0);
  for (int c = 0; c < nch; ++c) {
    if (c + 1 < nch) { stage((c + 1) & 1, (c + 1) * KC); async_wait_keep6(); }
    else             { async_wait_all(); }
    __syncthreads();                         // chunk c resident for all waves
    const __half* la = &smA[c & 1][(tmL * 16 + r) * KC];
    const __half* lb = &smB[c & 1][(tnL * 16 + r) * KC];
    #pragma unroll
    for (int kc = 0; kc < KC; kc += 32) {
      v16h av = load_a_frag(la + kc, hi);
      v16h bv = *(const v16h*)(lb + kc + hi * 16);
      acc = __builtin_amdgcn_wmma_f32_16x16x32_f16(false, av, false, bv,
                                                   (short)0, acc, false, false);
    }
    __syncthreads();                         // done reading before overwrite
  }
  const int orow = rowBase + tmL * 16;
  const int ocol = colBase + tnL * 16 + r;
  #pragma unroll
  for (int i = 0; i < 8; ++i)
    C[(size_t)(orow + i + 8 * hi) * N + ocol] = acc[i];   // row = i + 8*hi
}

// ------------------------------------------------------- weight f16 transpose
// W: [K,N] f32 -> Wt: [N,Kp] f16 (zero-pad K..Kp)
__global__ void k_convert_wt(const float* __restrict__ W, __half* __restrict__ Wt,
                             int K, int Kp, int N) {
  size_t idx = (size_t)blockIdx.x * TPB + threadIdx.x;
  if (idx >= (size_t)N * Kp) return;
  int kk = (int)(idx % Kp);
  int n  = (int)(idx / Kp);
  float v = (kk < K) ? W[(size_t)kk * N + n] : 0.0f;
  Wt[idx] = __float2half(v);
}

// ------------------------------------------------------------------- im2col
// conv1: x[2,128,128,3], 11x11 s2 SAME(pad lo 4) -> A[8192, 384] (K=363 padded)
__global__ void k_im2col_conv1(const float* __restrict__ x, __half* __restrict__ A) {
  int idx = blockIdx.x * TPB + threadIdx.x;
  if (idx >= 8192 * 384) return;
  int kk = idx % 384, m = idx / 384;
  int wo = m % 64, ho = (m / 64) % 64, b = m / 4096;
  float v = 0.0f;
  if (kk < 363) {
    int ci = kk % 3, rr = kk / 3;
    int kx = rr % 11, ky = rr / 11;
    int hi = ho * 2 - 4 + ky, wi = wo * 2 - 4 + kx;
    if (hi >= 0 && hi < 128 && wi >= 0 && wi < 128)
      v = x[((size_t)(b * 128 + hi) * 128 + wi) * 3 + ci];
  }
  A[idx] = __float2half(v);
}

// primarycaps: h[2,64,64,96], 9x9 s2 VALID -> A[1600, 7808] (M,K zero-padded)
__global__ void k_im2col_prim(const float* __restrict__ h, __half* __restrict__ A) {
  size_t idx = (size_t)blockIdx.x * TPB + threadIdx.x;
  if (idx >= (size_t)1600 * 7808) return;
  int kk = (int)(idx % 7808);
  int m  = (int)(idx / 7808);
  float v = 0.0f;
  if (m < 1568 && kk < 7776) {
    int wo = m % 28, ho = (m / 28) % 28, b = m / 784;
    int ci = kk % 96, rr = kk / 96;
    int kx = rr % 9, ky = rr / 9;
    v = h[((size_t)(b * 64 + ho * 2 + ky) * 64 + (wo * 2 + kx)) * 96 + ci];
  }
  A[idx] = __float2half(v);
}

// ---------------------------------------------------------------- epilogues
__global__ void k_ep_conv1(const float* __restrict__ C, const float* __restrict__ bias,
                           const float* __restrict__ bn, float* __restrict__ h) {
  int idx = blockIdx.x * TPB + threadIdx.x;
  if (idx >= 8192 * 96) return;
  int c = idx % 96;
  float v = fmaxf(C[idx] + bias[c], 0.0f);                     // relu
  h[idx] = (v - bn[192 + c]) * rsqrtf(bn[288 + c] + BNEPS) * bn[c] + bn[96 + c];
}

__global__ void k_ep_prim(const float* __restrict__ C, const float* __restrict__ pb,
                          const float* __restrict__ ab, const float* __restrict__ bn2,
                          float* __restrict__ poseA, float* __restrict__ actA) {
  int idx = blockIdx.x * TPB + threadIdx.x;
  if (idx >= 1568 * 544) return;
  int col = idx % 544, m = idx / 544;
  float v = C[idx];
  if (col < 512) {
    v += pb[col];
    int c = col >> 4;
    poseA[(size_t)m * 512 + col] =
        (v - bn2[64 + c]) * rsqrtf(bn2[96 + c] + BNEPS) * bn2[c] + bn2[32 + c];
  } else {
    int c = col - 512;
    v += ab[c];
    actA[(size_t)m * 32 + c] = 1.0f / (1.0f + __expf(-v));
  }
}

// in-place BN over pose [P,32,16], per-capsule (axis c)
__global__ void k_bn_caps(float* __restrict__ pose, const float* __restrict__ bn, int total) {
  int idx = blockIdx.x * TPB + threadIdx.x;
  if (idx >= total) return;
  int c = (idx >> 4) & 31;
  float v = pose[idx];
  pose[idx] = (v - bn[64 + c]) * rsqrtf(bn[96 + c] + BNEPS) * bn[c] + bn[32 + c];
}

// ------------------------------------------------------------------- votes
// votes[pl,n,o,i*4+k] = sum_j pose[..,n,i,j] * W[n,o,j,k], stored f16.
__global__ void k_votes(const float* __restrict__ pose, const float* __restrict__ Wv,
                        __half* __restrict__ votes, int posBase, int total,
                        int Nin, int Ho, int Wo, int Hi, int Wi, int convMode) {
  int idx = blockIdx.x * TPB + threadIdx.x;
  if (idx >= total) return;
  int d = idx & 15;
  int o = (idx >> 4) & 31;
  int rest = idx >> 9;
  int n = rest % Nin;
  int pl = rest / Nin;
  int pos = posBase + pl;
  int i = d >> 2, k = d & 3;
  size_t pbase;
  if (convMode) {
    int wo_ = pos % Wo, ho_ = (pos / Wo) % Ho, b = pos / (Wo * Ho);
    int p = n >> 5, c = n & 31;
    int ki = p / 3, kj = p % 3;
    pbase = ((size_t)((b * Hi + ho_ * 2 + ki) * Wi + (wo_ * 2 + kj)) * 32 + c) * 16 + i * 4;
  } else {
    pbase = ((size_t)pos * Nin + n) * 16 + i * 4;
  }
  const float* wp = Wv + ((size_t)n * 32 + o) * 16 + k;
  float acc = 0.0f;
  #pragma unroll
  for (int j = 0; j < 4; ++j) acc += pose[pbase + j] * wp[j * 4];
  votes[((size_t)pl * Nin + n) * 512 + o * 16 + d] = __float2half(acc);
}

// --------------------------------------------------------------- EM routing
// one workgroup (256 thr = 8 waves) per output position; 3 iterations.
// R in LDS (convcaps, Nin=288 -> 36KB) or global (dense, Nin=1152).
__global__ __launch_bounds__(TPB) void k_routing(
    const __half* __restrict__ votes, const float* __restrict__ act_in,
    float* __restrict__ pose_out, float* __restrict__ act_out,
    const float* __restrict__ beta_a, const float* __restrict__ beta_u,
    float* __restrict__ Rglob,
    int Nin, int posBase, int Ho, int Wo, int Hi, int Wi, int convMode) {
  extern __shared__ float sm[];
  float* base = sm;
  float* R;
  if (Rglob) { R = Rglob + (size_t)blockIdx.x * Nin * 32; }
  else       { R = base; base += Nin * 32; }
  float* ain  = base; base += Nin;
  float* mu   = base; base += 512;
  float* sig2 = base; base += 512;
  float* Rs   = base; base += 32;
  float* aout = base; base += 32;
  float* red  = base; base += 256;

  const int t = threadIdx.x;
  const int pl = blockIdx.x;
  const int pos = posBase + pl;
  const int wo_ = pos % Wo, ho_ = (pos / Wo) % Ho, b = pos / (Wo * Ho);

  for (int n = t; n < Nin; n += TPB) {
    float av;
    if (convMode) {
      int p = n >> 5, c = n & 31;
      int ki = p / 3, kj = p % 3;
      av = act_in[((size_t)((b * Hi + ho_ * 2 + ki) * Wi + (wo_ * 2 + kj))) * 32 + c];
    } else {
      av = act_in[(size_t)b * Nin + n];
    }
    ain[n] = av;
  }
  for (int i = t; i < Nin * 32; i += TPB) R[i] = 1.0f / 32.0f;
  __syncthreads();

  const __half* vbase = votes + (size_t)pl * Nin * 512;
  const int o = t & 31;
  const int part = t >> 5;      // 0..7
  const int d0 = part * 2;      // even d

  for (int it = 0; it < 3; ++it) {
    // Rs[o] = sum_n R*a + eps
    float ps = 0.0f;
    for (int n = part; n < Nin; n += 8) ps += R[n * 32 + o] * ain[n];
    red[part * 32 + o] = ps;
    __syncthreads();
    if (t < 32) {
      float s = 0.0f;
      for (int p = 0; p < 8; ++p) s += red[p * 32 + t];
      Rs[t] = s + CAPS_EPS;
    }
    __syncthreads();
    // mu[o,d] = sum_n R*a*votes / Rs
    {
      float a0 = 0.0f, a1 = 0.0f;
      for (int n = 0; n < Nin; ++n) {
        float ra = R[n * 32 + o] * ain[n];
        const __half* vp = vbase + ((size_t)n * 32 + o) * 16 + d0;
        a0 += ra * __half2float(vp[0]);
        a1 += ra * __half2float(vp[1]);
      }
      float inv = 1.0f / Rs[o];
      mu[o * 16 + d0] = a0 * inv;
      mu[o * 16 + d0 + 1] = a1 * inv;
    }
    __syncthreads();
    // sig2[o,d] = sum_n R*a*(v-mu)^2 / Rs + eps
    {
      float m0 = mu[o * 16 + d0], m1 = mu[o * 16 + d0 + 1];
      float a0 = 0.0f, a1 = 0.0f;
      for (int n = 0; n < Nin; ++n) {
        float ra = R[n * 32 + o] * ain[n];
        const __half* vp = vbase + ((size_t)n * 32 + o) * 16 + d0;
        float q0 = __half2float(vp[0]) - m0;
        float q1 = __half2float(vp[1]) - m1;
        a0 += ra * q0 * q0; a1 += ra * q1 * q1;
      }
      float inv = 1.0f / Rs[o];
      sig2[o * 16 + d0] = a0 * inv + CAPS_EPS;
      sig2[o * 16 + d0 + 1] = a1 * inv + CAPS_EPS;
    }
    __syncthreads();
    if (t < 32) {
      float cost = 0.0f;
      for (int d = 0; d < 16; ++d)
        cost += (beta_u[t] + 0.5f * __logf(sig2[t * 16 + d])) * Rs[t];
      float lam = 0.01f * (float)(it + 1);
      aout[t] = 1.0f / (1.0f + __expf(-lam * (beta_a[t] - cost)));
    }
    __syncthreads();
    if (it < 2) {
      for (int pr = t; pr < Nin * 32; pr += TPB) {
        int n = pr >> 5, oo = pr & 31;
        const __half* vp = vbase + ((size_t)n * 32 + oo) * 16;
        float lp = 0.0f;
        for (int d = 0; d < 16; ++d) {
          float s2 = sig2[oo * 16 + d];
          float df = __half2float(vp[d]) - mu[oo * 16 + d];
          lp += df * df / s2 + __logf(6.2831853071f * s2);
        }
        R[pr] = __logf(aout[oo] + CAPS_EPS) - 0.5f * lp;   // logits (temp)
      }
      __syncthreads();
      for (int n = t; n < Nin; n += TPB) {                 // softmax over o
        float mx = -1e30f;
        for (int oo = 0; oo < 32; ++oo) mx = fmaxf(mx, R[n * 32 + oo]);
        float s = 0.0f;
        for (int oo = 0; oo < 32; ++oo) {
          float e = __expf(R[n * 32 + oo] - mx);
          R[n * 32 + oo] = e; s += e;
        }
        float inv = 1.0f / s;
        for (int oo = 0; oo < 32; ++oo) R[n * 32 + oo] *= inv;
      }
      __syncthreads();
    }
  }
  for (int i = t; i < 512; i += TPB) pose_out[(size_t)pos * 512 + i] = mu[i];
  if (t < 32) act_out[(size_t)pos * 32 + t] = aout[t];
}

// -------------------------------------------------------------- decoder glue
__global__ void k_dec_prepare(const float* __restrict__ src, __half* __restrict__ A, int K) {
  int idx = blockIdx.x * TPB + threadIdx.x;
  if (idx >= 64 * K) return;
  int r = idx / K;
  A[idx] = __float2half(r < 2 ? src[idx] : 0.0f);
}

__global__ void k_dec_relu(const float* __restrict__ C, const float* __restrict__ bias,
                           __half* __restrict__ A, int N) {
  int idx = blockIdx.x * TPB + threadIdx.x;
  if (idx >= 64 * N) return;
  int r = idx / N, c = idx % N;
  float v = (r < 2) ? fmaxf(C[idx] + bias[c], 0.0f) : 0.0f;
  A[idx] = __float2half(v);
}

__global__ void k_dec_final(const float* __restrict__ C, const float* __restrict__ bias,
                            float* __restrict__ out, int N) {
  int idx = blockIdx.x * TPB + threadIdx.x;
  if (idx >= 2 * N) return;
  out[idx] = C[idx] + bias[idx % N];
}

// ================================================================== host
static inline unsigned gcount(size_t n) { return (unsigned)((n + TPB - 1) / TPB); }

extern "C" void kernel_launch(void* const* d_in, const int* in_sizes, int n_in,
                              void* d_out, int out_size, void* d_ws, size_t ws_size,
                              hipStream_t stream) {
  (void)in_sizes; (void)n_in; (void)out_size; (void)ws_size;
  const float* x           = (const float*)d_in[0];
  const float* conv_w      = (const float*)d_in[1];
  const float* conv_b      = (const float*)d_in[2];
  const float* bn1         = (const float*)d_in[3];
  const float* prim_pose_w = (const float*)d_in[4];
  const float* prim_pose_b = (const float*)d_in[5];
  const float* prim_act_w  = (const float*)d_in[6];
  const float* prim_act_b  = (const float*)d_in[7];
  const float* bn2         = (const float*)d_in[8];
  const float* cc1_w       = (const float*)d_in[9];
  const float* cc1_ba      = (const float*)d_in[10];
  const float* cc1_bu      = (const float*)d_in[11];
  const float* bn3         = (const float*)d_in[12];
  const float* cc2_w       = (const float*)d_in[13];
  const float* cc2_ba      = (const float*)d_in[14];
  const float* cc2_bu      = (const float*)d_in[15];
  const float* bn4         = (const float*)d_in[16];
  const float* dc_w        = (const float*)d_in[17];
  const float* dc_ba       = (const float*)d_in[18];
  const float* dc_bu       = (const float*)d_in[19];
  const float* dec_w1      = (const float*)d_in[20];
  const float* dec_b1      = (const float*)d_in[21];
  const float* dec_w2      = (const float*)d_in[22];
  const float* dec_b2      = (const float*)d_in[23];
  const float* dec_w3      = (const float*)d_in[24];
  const float* dec_b3      = (const float*)d_in[25];

  // ---- workspace partition (~72 MB) ----
  char* wp = (char*)d_ws;
  auto alloc = [&](size_t bytes) {
    char* p = wp; wp += (bytes + 255) & ~(size_t)255; return p;
  };
  float*  h_buf   = (float*)alloc((size_t)8192 * 96 * 4);
  float*  poseA   = (float*)alloc((size_t)1568 * 512 * 4);
  float*  actA    = (float*)alloc((size_t)1568 * 32 * 4);
  float*  pose1   = (float*)alloc((size_t)338 * 512 * 4);
  float*  act1    = (float*)alloc((size_t)338 * 32 * 4);
  float*  pose2   = (float*)alloc((size_t)72 * 512 * 4);
  float*  act2    = (float*)alloc((size_t)72 * 32 * 4);
  float*  mu_d    = (float*)alloc((size_t)2 * 512 * 4);
  float*  act_d   = (float*)alloc((size_t)2 * 32 * 4);
  float*  Rg      = (float*)alloc((size_t)2 * 1152 * 32 * 4);     // dense R
  __half* Af16    = (__half*)alloc((size_t)1600 * 7808 * 2);      // max im2col
  __half* Wf16    = (__half*)alloc((size_t)8192 * 1024 * 2);      // max Wt
  float*  Cbuf    = (float*)alloc((size_t)1600 * 544 * 4);        // max GEMM C
  __half* votesB  = (__half*)alloc((size_t)64 * 288 * 512 * 2);   // votes chunk

  auto gemm = [&](const __half* A, const __half* Bt, float* C, int M, int N, int K) {
    k_gemm_wmma<<<dim3((unsigned)((M / 64) * (N / 32))), dim3(TPB), 0, stream>>>(A, Bt, C, M, N, K);
  };

  // ---- conv1: im2col GEMM [8192,384]x[384,96] + relu + bn1 ----
  k_convert_wt<<<gcount((size_t)96 * 384), TPB, 0, stream>>>(conv_w, Wf16, 363, 384, 96);
  k_im2col_conv1<<<gcount((size_t)8192 * 384), TPB, 0, stream>>>(x, Af16);
  gemm(Af16, Wf16, Cbuf, 8192, 96, 384);
  k_ep_conv1<<<gcount((size_t)8192 * 96), TPB, 0, stream>>>(Cbuf, conv_b, bn1, h_buf);

  // ---- primarycaps: GEMM [1600,7808]x[7808,544]; pose bn2 / act sigmoid ----
  k_convert_wt<<<gcount((size_t)512 * 7808), TPB, 0, stream>>>(prim_pose_w, Wf16, 7776, 7808, 512);
  k_convert_wt<<<gcount((size_t)32 * 7808), TPB, 0, stream>>>(prim_act_w, Wf16 + (size_t)512 * 7808, 7776, 7808, 32);
  k_im2col_prim<<<gcount((size_t)1600 * 7808), TPB, 0, stream>>>(h_buf, Af16);
  gemm(Af16, Wf16, Cbuf, 1600, 544, 7808);
  k_ep_prim<<<gcount((size_t)1568 * 544), TPB, 0, stream>>>(Cbuf, prim_pose_b, prim_act_b, bn2, poseA, actA);

  // ---- convcaps layers (chunked votes + EM routing) ----
  auto run_caps = [&](const float* poseIn, const float* actIn, const float* Wv,
                      const float* ba, const float* bu, float* poseOut, float* actOut,
                      int Ho, int Wo, int Hi, int Wi) {
    const int P = 2 * Ho * Wo;
    const size_t smem = ((size_t)288 * 33 + 1344) * 4;   // R in LDS
    for (int b0 = 0; b0 < P; b0 += 64) {
      int cp = P - b0; if (cp > 64) cp = 64;
      int tot = cp * 288 * 512;
      k_votes<<<gcount((size_t)tot), TPB, 0, stream>>>(poseIn, Wv, votesB, b0, tot,
                                                       288, Ho, Wo, Hi, Wi, 1);
      k_routing<<<dim3(cp), dim3(TPB), smem, stream>>>(votesB, actIn, poseOut, actOut,
                                                       ba, bu, nullptr, 288, b0,
                                                       Ho, Wo, Hi, Wi, 1);
    }
  };
  run_caps(poseA, actA, cc1_w, cc1_ba, cc1_bu, pose1, act1, 13, 13, 28, 28);
  k_bn_caps<<<gcount((size_t)338 * 512), TPB, 0, stream>>>(pose1, bn3, 338 * 512);
  run_caps(pose1, act1, cc2_w, cc2_ba, cc2_bu, pose2, act2, 6, 6, 13, 13);
  k_bn_caps<<<gcount((size_t)72 * 512), TPB, 0, stream>>>(pose2, bn4, 72 * 512);

  // ---- dense caps: Nin=1152, 2 positions, R in global scratch ----
  {
    int tot = 2 * 1152 * 512;
    k_votes<<<gcount((size_t)tot), TPB, 0, stream>>>(pose2, dc_w, votesB, 0, tot,
                                                     1152, 1, 1, 0, 0, 0);
    const size_t smem = ((size_t)1152 + 1344) * 4;
    k_routing<<<dim3(2), dim3(TPB), smem, stream>>>(votesB, act2, mu_d, act_d,
                                                    dc_ba, dc_bu, Rg, 1152, 0,
                                                    1, 1, 0, 0, 0);
  }

  // ---- decoder MLP (M padded 2->64 for the 64-row macro-tile GEMM) ----
  k_convert_wt<<<gcount((size_t)512 * 512), TPB, 0, stream>>>(dec_w1, Wf16, 512, 512, 512);
  k_dec_prepare<<<gcount((size_t)64 * 512), TPB, 0, stream>>>(mu_d, Af16, 512);
  gemm(Af16, Wf16, Cbuf, 64, 512, 512);
  k_dec_relu<<<gcount((size_t)64 * 512), TPB, 0, stream>>>(Cbuf, dec_b1, Af16, 512);

  k_convert_wt<<<gcount((size_t)1024 * 512), TPB, 0, stream>>>(dec_w2, Wf16, 512, 512, 1024);
  gemm(Af16, Wf16, Cbuf, 64, 1024, 512);
  k_dec_relu<<<gcount((size_t)64 * 1024), TPB, 0, stream>>>(Cbuf, dec_b2, Af16, 1024);

  k_convert_wt<<<gcount((size_t)8192 * 1024), TPB, 0, stream>>>(dec_w3, Wf16, 1024, 1024, 8192);
  gemm(Af16, Wf16, Cbuf, 64, 8192, 1024);
  k_dec_final<<<gcount((size_t)2 * 8192), TPB, 0, stream>>>(Cbuf, dec_b3, (float*)d_out, 8192);
}
